// MultiHeadAttention_87952340287565
// MI455X (gfx1250) — compile-verified, matrix operand order checked by reading, not run on hardware
//
#include <hip/hip_runtime.h>
#include <math.h>

// ---------------------------------------------------------------------------
// MI455X (gfx1250) implementation of grouped-projection MHA + gated output.
// GEMMs on v_wmma_f32_16x16x32_bf16 (f32 -> bf16 at tile staging / fragment
// gather; f32 accumulation). Projection GEMM A-tiles staged with the CDNA5
// async global->LDS path (GLOBAL_LOAD_ASYNC_TO_LDS_B128 + s_wait_asynccnt).
// ---------------------------------------------------------------------------

typedef __bf16 bf16_t;
typedef __attribute__((ext_vector_type(16))) __bf16        v16bf;
typedef __attribute__((ext_vector_type(8)))  float         v8f;
typedef __attribute__((ext_vector_type(4)))  unsigned int  v4u;

union FragCast { v4u u[2]; v16bf f; };

// Pack two f32 into two bf16 in one dword (LDS store as b32).
__device__ __forceinline__ unsigned pack_bf16x2(float lo, float hi) {
    union { bf16_t b[2]; unsigned u; } t;
    t.b[0] = (bf16_t)lo; t.b[1] = (bf16_t)hi;
    return t.u;
}

// Low 32 bits of a generic pointer to LDS = wave-relative LDS byte offset
// (ISA: LDS aperture keeps the offset in addr[31:0]).
__device__ __forceinline__ unsigned lds_addr_of(const void* p) {
    return (unsigned)(unsigned long long)p;
}

// CDNA5 async copy: 16 bytes global -> LDS, tracked by ASYNCcnt.
// GVS mode: uniform 64-bit SGPR base + per-lane signed 32-bit byte offset.
__device__ __forceinline__ void async_copy16(unsigned lds_off,
                                             const float* base,
                                             unsigned byte_off) {
    asm volatile("global_load_async_to_lds_b128 %0, %1, %2"
                 :: "v"(lds_off), "v"(byte_off), "s"(base)
                 : "memory");
}
__device__ __forceinline__ void wait_async0() {
    asm volatile("s_wait_asynccnt 0" ::: "memory");
}

// Gather a 16x32 bf16 WMMA operand fragment from a bf16 LDS tile (row-major,
// leading dim `ld` elements). Per CDNA5 ISA 16-bit A layout: lanes 0-15 hold
// row M=lane with K = k0+0..7 and k0+16..23; lanes 16-31 hold K shifted by 8.
__device__ __forceinline__ v16bf load_frag(const bf16_t* tile, int ld,
                                           int row0, int k0, int lane) {
    const int r  = row0 + (lane & 15);
    const int kb = k0 + ((lane >> 4) << 3);   // +0 or +8
    FragCast fc;
    fc.u[0] = *reinterpret_cast<const v4u*>(tile + r * ld + kb);
    fc.u[1] = *reinterpret_cast<const v4u*>(tile + r * ld + kb + 16);
    return fc.f;
}

// Same fragment gather but from an f32 LDS tile (async-copied raw), with
// f32 -> bf16 conversion in registers.
__device__ __forceinline__ v16bf load_frag_f32(const float* tile, int ld,
                                               int row0, int k0, int lane) {
    const int r  = row0 + (lane & 15);
    const int kb = k0 + ((lane >> 4) << 3);
    const float* p = tile + r * ld + kb;
    v16bf f;
#pragma unroll
    for (int i = 0; i < 8; ++i) f[i] = (bf16_t)p[i];
#pragma unroll
    for (int i = 0; i < 8; ++i) f[i + 8] = (bf16_t)p[i + 16];
    return f;
}

__device__ __forceinline__ v8f wmma_bf16(v16bf a, v16bf b, v8f c) {
    // (neg_a, A, neg_b, B, c_mod, C, reuse_a, reuse_b)
    return __builtin_amdgcn_wmma_f32_16x16x32_bf16(false, a, false, b,
                                                   (short)0, c, false, false);
}

// ---------------------------------------------------------------------------
// Generic batched GEMM, 64x64 output tile per block, 4 waves (wave32),
// each wave owns a 32x32 quadrant (2x2 WMMA accumulators).
//   C[b] = A[b] (MxK, f32) * B[b] (KxN if TRANSB else stored as [N][K])
// A tile staged via async global->LDS (raw f32); B tile converted to bf16
// while staging (transposed for TRANSB).
// ---------------------------------------------------------------------------
template <bool TRANSB>
__global__ __launch_bounds__(128) void gemm_tile64(
    const float* __restrict__ A, long lda, long aBatch,
    const float* __restrict__ B, long ldb, long bBatch,
    float* __restrict__ C, long ldc, long cBatch,
    int K, int mTiles)
{
    __shared__ __align__(16) float  Af[64 * 32];   // [m][k] raw f32 (async dest)
    __shared__ __align__(16) bf16_t Bs[64 * 32];   // [n][k] bf16

    const int tid  = threadIdx.x;
    const int lane = tid & 31;
    const int wave = tid >> 5;
    const int wm   = (wave & 1) << 5;
    const int wn   = (wave >> 1) << 5;
    const int tm   = (int)blockIdx.x % mTiles;
    const int tn   = (int)blockIdx.x / mTiles;

    const float* Ab = A + (long)blockIdx.y * aBatch + (long)tm * 64 * lda;
    const float* Bb = B + (long)blockIdx.y * bBatch;
    float*       Cb = C + (long)blockIdx.y * cBatch + (long)tm * 64 * ldc + (long)tn * 64;

    v8f acc00 = {}, acc01 = {}, acc10 = {}, acc11 = {};

    for (int k0 = 0; k0 < K; k0 += 32) {
        // --- stage A tile 64x32 f32 via async global->LDS (512 x 16B) ---
        {
            const float* srcBase = Ab + k0;         // block-uniform -> SGPRs
            for (int idx = tid; idx < 512; idx += 128) {
                const int r  = idx >> 3;
                const int c4 = (idx & 7) << 2;
                const unsigned goff = (unsigned)(((long)r * lda + c4) * 4);
                async_copy16(lds_addr_of(&Af[r * 32 + c4]), srcBase, goff);
            }
        }
        // --- stage B tile as Bs[n][k] (f32 -> bf16) ---
        if (TRANSB) {  // B is [K][N] row-major: transpose while staging
            for (int idx = tid; idx < 2048; idx += 128) {
                const int n  = idx & 63;            // coalesced along n
                const int kk = idx >> 6;
                Bs[n * 32 + kk] = (bf16_t)Bb[(long)(k0 + kk) * ldb + tn * 64 + n];
            }
        } else {       // B already stored as [N][K] row-major
            for (int idx = tid; idx < 512; idx += 128) {
                const int n = idx >> 3;
                const int c = (idx & 7) << 2;
                const float4 b4 = *reinterpret_cast<const float4*>(Bb + (long)(tn * 64 + n) * ldb + k0 + c);
                unsigned* dst = reinterpret_cast<unsigned*>(&Bs[n * 32 + c]);
                dst[0] = pack_bf16x2(b4.x, b4.y);
                dst[1] = pack_bf16x2(b4.z, b4.w);
            }
        }
        // prefetch next B k-rows (the HBM-bound weight stream)
        if (k0 + 32 < K)
            __builtin_prefetch(Bb + (long)(k0 + 32 + (tid & 31)) * ldb + tn * 64 +
                                   ((tid >> 5) << 4), 0, 1);

        wait_async0();       // this wave's async copies have landed in LDS
        __syncthreads();     // all waves' copies + B staging visible

        const v16bf a0 = load_frag_f32(Af, 32, wm,      0, lane);
        const v16bf a1 = load_frag_f32(Af, 32, wm + 16, 0, lane);
        const v16bf b0 = load_frag(Bs, 32, wn,      0, lane);
        const v16bf b1 = load_frag(Bs, 32, wn + 16, 0, lane);
        acc00 = wmma_bf16(a0, b0, acc00);
        acc01 = wmma_bf16(a0, b1, acc01);
        acc10 = wmma_bf16(a1, b0, acc10);
        acc11 = wmma_bf16(a1, b1, acc11);

        __syncthreads();
    }

    // Epilogue: C/D layout — VGPR i, lanes 0-15 row M=i, lanes 16-31 row M=i+8.
    const int cn = lane & 15;
    const int hi = (lane >> 4) << 3;
    for (int r = 0; r < 8; ++r) {
        Cb[(long)(wm +      hi + r) * ldc + wn +      cn] = acc00[r];
        Cb[(long)(wm +      hi + r) * ldc + wn + 16 + cn] = acc01[r];
        Cb[(long)(wm + 16 + hi + r) * ldc + wn +      cn] = acc10[r];
        Cb[(long)(wm + 16 + hi + r) * ldc + wn + 16 + cn] = acc11[r];
    }
}

// ---------------------------------------------------------------------------
// Fused attention: one block per (b, h). 64x64 scores (WMMA), f32 softmax in
// LDS (also emits the attn output tensor), then P @ V (WMMA).
// qp/kp/vp are f32 [B, N, H*64]; attn f32 [B,H,64,64]; ao f32 [B, 64, 1024].
// ---------------------------------------------------------------------------
__global__ __launch_bounds__(128) void attention_kernel(
    const float* __restrict__ qp, const float* __restrict__ kp,
    const float* __restrict__ vp, float* __restrict__ attn,
    float* __restrict__ ao)
{
    const int b = (int)blockIdx.x >> 4;
    const int h = (int)blockIdx.x & 15;

    __shared__ __align__(16) bf16_t Qs[64 * 64];    // [q][d]
    __shared__ __align__(16) bf16_t Ks[64 * 64];    // [k][d]  (Bt layout for QK^T)
    __shared__ __align__(16) bf16_t VTs[64 * 64];   // [d][k]  (Bt layout for P@V)
    __shared__ __align__(16) bf16_t Ps[64 * 64];    // attn probs bf16
    __shared__ float Ss[64 * 64];                   // scores / probs f32

    const int tid  = threadIdx.x;
    const int lane = tid & 31;
    const int wave = tid >> 5;

    for (int idx = tid; idx < 4096; idx += 128) {
        const int n = idx >> 6;
        const int d = idx & 63;
        const long base = ((long)b * 64 + n) * 1024 + h * 64 + d;
        Qs[n * 64 + d]  = (bf16_t)qp[base];
        Ks[n * 64 + d]  = (bf16_t)kp[base];
        VTs[d * 64 + n] = (bf16_t)vp[base];
    }
    __syncthreads();

    // scores: wave owns 16 query rows x all 64 key cols
    v8f sacc[4] = {{}, {}, {}, {}};
    for (int k0 = 0; k0 < 64; k0 += 32) {
        const v16bf a = load_frag(Qs, 64, wave * 16, k0, lane);
        for (int j = 0; j < 4; ++j) {
            const v16bf bb = load_frag(Ks, 64, j * 16, k0, lane);
            sacc[j] = wmma_bf16(a, bb, sacc[j]);
        }
    }
    const int cn = lane & 15;
    const int hi = (lane >> 4) << 3;
    const float inv_temp = 0.125f;  // 1/sqrt(DK=64)
    for (int j = 0; j < 4; ++j)
        for (int r = 0; r < 8; ++r)
            Ss[(wave * 16 + hi + r) * 64 + j * 16 + cn] = sacc[j][r] * inv_temp;
    __syncthreads();

    // softmax per row; emit attn (f32, global output) and Ps (bf16 in LDS)
    if (tid < 64) {
        float mx = -1e30f;
        for (int c = 0; c < 64; ++c) mx = fmaxf(mx, Ss[tid * 64 + c]);
        float sum = 0.f;
        for (int c = 0; c < 64; ++c) {
            const float e = __expf(Ss[tid * 64 + c] - mx);
            Ss[tid * 64 + c] = e;
            sum += e;
        }
        const float inv = 1.f / sum;
        float* arow = attn + (((long)b * 16 + h) * 64 + tid) * 64;
        for (int c = 0; c < 64; ++c) {
            const float p = Ss[tid * 64 + c] * inv;
            arow[c] = p;
            Ps[tid * 64 + c] = (bf16_t)p;
        }
    }
    __syncthreads();

    // out = P @ V : wave owns 16 rows x 64 cols; Bt tile is VTs[d][k]
    v8f oacc[4] = {{}, {}, {}, {}};
    for (int k0 = 0; k0 < 64; k0 += 32) {
        const v16bf a = load_frag(Ps, 64, wave * 16, k0, lane);
        for (int j = 0; j < 4; ++j) {
            const v16bf bb = load_frag(VTs, 64, j * 16, k0, lane);
            oacc[j] = wmma_bf16(a, bb, oacc[j]);
        }
    }
    for (int j = 0; j < 4; ++j)
        for (int r = 0; r < 8; ++r) {
            const int qrow = wave * 16 + hi + r;
            const int d    = j * 16 + cn;
            ao[((long)b * 64 + qrow) * 1024 + h * 64 + d] = oacc[j][r];
        }
}

// ---------------------------------------------------------------------------
// Fused fc + gate: out = sigmoid(X@gw^T + gb) * tanh(X@fcw^T + fcb)
// X f32 [8192,1024]; fcw/gw f32 [1024,1024] already in [N][K] (Bt) layout.
// ---------------------------------------------------------------------------
__global__ __launch_bounds__(128) void fc_gate_kernel(
    const float* __restrict__ X,
    const float* __restrict__ fcw, const float* __restrict__ fcb,
    const float* __restrict__ gw,  const float* __restrict__ gb,
    float* __restrict__ out)
{
    __shared__ __align__(16) bf16_t Xs[64 * 32];
    __shared__ __align__(16) bf16_t Fs[64 * 32];
    __shared__ __align__(16) bf16_t Gs[64 * 32];

    const int tid  = threadIdx.x;
    const int lane = tid & 31;
    const int wave = tid >> 5;
    const int wm   = (wave & 1) << 5;
    const int wn   = (wave >> 1) << 5;
    const int tm   = (int)blockIdx.x & 127;
    const int tn   = (int)blockIdx.x >> 7;

    const float* Xb = X + (long)tm * 64 * 1024;

    v8f f00 = {}, f01 = {}, f10 = {}, f11 = {};
    v8f g00 = {}, g01 = {}, g10 = {}, g11 = {};

    for (int k0 = 0; k0 < 1024; k0 += 32) {
        for (int idx = tid; idx < 512; idx += 128) {
            const int r = idx >> 3;
            const int c = (idx & 7) << 2;
            const float4 xa = *reinterpret_cast<const float4*>(Xb + (long)r * 1024 + k0 + c);
            unsigned* d0 = reinterpret_cast<unsigned*>(&Xs[r * 32 + c]);
            d0[0] = pack_bf16x2(xa.x, xa.y); d0[1] = pack_bf16x2(xa.z, xa.w);
            const float4 fa = *reinterpret_cast<const float4*>(fcw + (long)(tn * 64 + r) * 1024 + k0 + c);
            unsigned* d1 = reinterpret_cast<unsigned*>(&Fs[r * 32 + c]);
            d1[0] = pack_bf16x2(fa.x, fa.y); d1[1] = pack_bf16x2(fa.z, fa.w);
            const float4 ga = *reinterpret_cast<const float4*>(gw + (long)(tn * 64 + r) * 1024 + k0 + c);
            unsigned* d2 = reinterpret_cast<unsigned*>(&Gs[r * 32 + c]);
            d2[0] = pack_bf16x2(ga.x, ga.y); d2[1] = pack_bf16x2(ga.z, ga.w);
        }
        if (k0 + 32 < 1024)
            __builtin_prefetch(Xb + (long)(tid & 63) * 1024 + k0 + 32, 0, 1);
        __syncthreads();

        const v16bf a0 = load_frag(Xs, 32, wm,      0, lane);
        const v16bf a1 = load_frag(Xs, 32, wm + 16, 0, lane);
        const v16bf fb0 = load_frag(Fs, 32, wn,      0, lane);
        const v16bf fb1 = load_frag(Fs, 32, wn + 16, 0, lane);
        const v16bf gb0 = load_frag(Gs, 32, wn,      0, lane);
        const v16bf gb1 = load_frag(Gs, 32, wn + 16, 0, lane);
        f00 = wmma_bf16(a0, fb0, f00);
        f01 = wmma_bf16(a0, fb1, f01);
        f10 = wmma_bf16(a1, fb0, f10);
        f11 = wmma_bf16(a1, fb1, f11);
        g00 = wmma_bf16(a0, gb0, g00);
        g01 = wmma_bf16(a0, gb1, g01);
        g10 = wmma_bf16(a1, gb0, g10);
        g11 = wmma_bf16(a1, gb1, g11);

        __syncthreads();
    }

    const int cn = lane & 15;
    const int hi = (lane >> 4) << 3;
    const v8f* facc[2][2] = {{&f00, &f01}, {&f10, &f11}};
    const v8f* gacc[2][2] = {{&g00, &g01}, {&g10, &g11}};
    for (int i = 0; i < 2; ++i)
        for (int j = 0; j < 2; ++j)
            for (int r = 0; r < 8; ++r) {
                const int row = tm * 64 + wm + i * 16 + hi + r;
                const int col = tn * 64 + wn + j * 16 + cn;
                const float fv = (*facc[i][j])[r] + fcb[col];
                const float gv = (*gacc[i][j])[r] + gb[col];
                const float sig = 1.f / (1.f + __expf(-gv));
                out[(long)row * 1024 + col] = sig * tanhf(fv);
            }
}

// ---------------------------------------------------------------------------
extern "C" void kernel_launch(void* const* d_in, const int* in_sizes, int n_in,
                              void* d_out, int out_size, void* d_ws, size_t ws_size,
                              hipStream_t stream)
{
    (void)in_sizes; (void)n_in; (void)out_size; (void)ws_size;

    const float* q   = (const float*)d_in[0];
    const float* k   = (const float*)d_in[1];
    const float* v   = (const float*)d_in[2];
    const float* w_q = (const float*)d_in[3];
    const float* w_k = (const float*)d_in[4];
    const float* w_v = (const float*)d_in[5];
    const float* fcw = (const float*)d_in[6];
    const float* fcb = (const float*)d_in[7];
    const float* gw  = (const float*)d_in[8];
    const float* gb  = (const float*)d_in[9];
    float* out = (float*)d_out;               // [output | attn] concatenated

    const long PROJ = 128L * 64 * 1024;       // 8,388,608 elements per tensor
    float* qp = (float*)d_ws;                 // ws: 4 x 32 MiB f32 = 128 MiB
    float* kp = qp + PROJ;
    float* vp = kp + PROJ;
    float* ao = vp + PROJ;

    const dim3 blk(128);
    // Grouped projections: batch = position n (64); per-n A offset 1024,
    // per-n weight offset 1024*1024. M=128 (2 m-tiles), N=1024, K=1024.
    const dim3 gp(32, 64);
    gemm_tile64<true><<<gp, blk, 0, stream>>>(q, 65536, 1024, w_q, 1024, 1048576,
                                              qp, 65536, 1024, 1024, 2);
    gemm_tile64<true><<<gp, blk, 0, stream>>>(k, 65536, 1024, w_k, 1024, 1048576,
                                              kp, 65536, 1024, 1024, 2);
    gemm_tile64<true><<<gp, blk, 0, stream>>>(v, 65536, 1024, w_v, 1024, 1048576,
                                              vp, 65536, 1024, 1024, 2);
    // Attention: one block per (b,h); writes attn to out[PROJ..] and ao to ws.
    attention_kernel<<<dim3(128 * 16), blk, 0, stream>>>(qp, kp, vp, out + PROJ, ao);
    // Fused fc + gate + sigmoid*tanh into out[0..PROJ).
    fc_gate_kernel<<<dim3(128 * 16), blk, 0, stream>>>(ao, fcw, fcb, gw, gb, out);
}
